// Community2Emb_74758200754585
// MI455X (gfx1250) — compile-verified, matrix operand order checked by reading, not run on hardware
//
#include <hip/hip_runtime.h>
#include <hip/hip_bf16.h>

// ---------------------------------------------------------------------------
// Community2Emb GMM loss for MI455X (gfx1250)
//   x = embedding[labels]  (B=65536, D=64)
//   quad[b,k] = (x-mu_k)^T invcov_k (x-mu_k),  K=32
//   loss = | sum_{b,k} pi[labels[b],k] * (-0.5)(D log2pi + logdet_k + quad) |
//
// Expansion: quad = x^T M x - 2 (M mu)·x + (mu^T M mu), so the WMMA A-operand
// (split-bf16 of x) is k-invariant and built once per wave.  Kernel 1 inverts
// cov, and writes M_k as bf16 hi/lo ALREADY swizzled into the WMMA B-fragment
// layout; the main loop's Tensor Data Mover streams that 16KB block per k
// (double buffered, s_wait_tensorcnt) so fragments load as ds_load_b128 with
// zero per-k conversion work.  Split-bf16 3-product (h*h + h*l + l*h, f32
// accumulate) gives ~f32 accuracy at the bf16 matrix-pipe rate.
// ---------------------------------------------------------------------------

typedef float  v8f   __attribute__((ext_vector_type(8)));
typedef __bf16 v16bf __attribute__((ext_vector_type(16)));
typedef unsigned int u32x4 __attribute__((ext_vector_type(4)));
typedef int    i32x4 __attribute__((ext_vector_type(4)));
typedef int    i32x8 __attribute__((ext_vector_type(8)));

#define KCOMP 32
#define DDIM  64
#define TROWS 128   // rows per workgroup (8 waves x 16-row WMMA tiles)

// 16-bit WMMA fragment K-index for component `comp` (0..15), lane-half `h2`:
//   VGPR v = comp/2 holds K = {2v,2v+1} (v<4) or {16+2(v-4), ...} (v>=4),
//   lanes 16-31 add 8.  (ISA 7.12.2, 16-bit A 16x32 table; B mirrored.)
__device__ __forceinline__ int frag_k(int comp, int h2) {
  int v = comp >> 1, odd = comp & 1;
  return ((v < 4) ? (2 * v) : (16 + 2 * (v - 4))) + odd + 8 * h2;
}

// ---- TDM: load one 16KB contiguous block (as 64x64 of 4B) global -> LDS ----
__device__ __forceinline__ void tdm_load_16k(unsigned lds_byte_addr,
                                             unsigned long long gaddr) {
  u32x4 g0;
  g0[0] = 1u;                                   // count=1, user mode, no gather
  g0[1] = lds_byte_addr;                        // lds_addr
  g0[2] = (unsigned)(gaddr & 0xFFFFFFFFull);    // global_addr[31:0]
  g0[3] = (unsigned)((gaddr >> 32) & 0x1FFFFFFull) | (2u << 30); // [56:32], type=2

  i32x8 g1;
  g1[0] = 0x00020000;        // workgroup_mask=0, data_size=2 (4B), no pad/iter
  g1[1] = (int)(64u << 16);  // tensor_dim0[15:0] = 64
  g1[2] = (int)(64u << 16);  // tensor_dim1[15:0] = 64
  g1[3] = (int)(64u << 16);  // tile_dim0 = 64
  g1[4] = 64;                // tile_dim1 = 64, tile_dim2 = 0
  g1[5] = 64;                // tensor_dim0_stride[31:0] = 64
  g1[6] = 0;
  g1[7] = 0;

  i32x4 z4 = {0, 0, 0, 0};
#if defined(__clang_major__) && (__clang_major__ >= 23)
  i32x8 z8 = {0, 0, 0, 0, 0, 0, 0, 0};
  __builtin_amdgcn_tensor_load_to_lds(g0, g1, z4, z4, z8, 0);
#else
  __builtin_amdgcn_tensor_load_to_lds(g0, g1, z4, z4, 0);
#endif
}

// ---- Kernel 0: zero the global accumulator --------------------------------
__global__ void zero_acc_kernel(float* acc) { acc[0] = 0.0f; }

// ---- Kernel 1: invert cov_k (Gauss-Jordan), emit:
//   * bf16 hi/lo of M_k in B-fragment order (16KB/k, Bh[4096] then Bl[4096])
//   * v_k = M_k mu_k (f32, 64)
//   * w_k = mu_k^T M_k mu_k + logdet_k + D*log(2*pi)
__global__ __launch_bounds__(256) void invert_prep_kernel(
    const float* __restrict__ cov, const float* __restrict__ cen,
    __bf16* __restrict__ wsB, float* __restrict__ vks,
    float* __restrict__ wk) {
  constexpr int S = 132;  // padded row stride for 128 augmented columns
  __shared__ float aug[DDIM * S];
  __shared__ float fac[DDIM];
  __shared__ float vsh[DDIM];
  __shared__ float ldacc;

  const int k = blockIdx.x;
  const int tid = threadIdx.x;
  const float* A = cov + (size_t)k * DDIM * DDIM;
  const float* mu = cen + (size_t)k * DDIM;

  for (int i = tid; i < DDIM * DDIM; i += 256) {
    int r = i >> 6, c = i & 63;
    aug[r * S + c] = A[i];
    aug[r * S + 64 + c] = (r == c) ? 1.0f : 0.0f;
  }
  if (tid == 0) ldacc = 0.0f;
  __syncthreads();

  for (int p = 0; p < DDIM; ++p) {
    float piv = aug[p * S + p];          // read by all before it is scaled
    float rp = 1.0f / piv;
    if (tid == 0) ldacc += logf(piv);    // SPD -> pivots positive
    __syncthreads();
    if (tid < 128) aug[p * S + tid] *= rp;
    __syncthreads();
    if (tid < DDIM) fac[tid] = aug[tid * S + p];
    __syncthreads();
    for (int i = tid; i < DDIM * 128; i += 256) {
      int r = i >> 7, c = i & 127;
      if (r != p) aug[r * S + c] -= fac[r] * aug[p * S + c];
    }
    __syncthreads();
  }

  // split-bf16 of M in B-fragment order
  __bf16* dst = wsB + (size_t)k * 8192;
  for (int i = tid; i < 4096; i += 256) {
    int c = i >> 11, nt = (i >> 9) & 3, ln = (i >> 4) & 31, comp = i & 15;
    int kk = c * 32 + frag_k(comp, ln >> 4);
    int n = nt * 16 + (ln & 15);
    float f = aug[kk * S + 64 + n];      // M[kk][n]
    __bf16 h = (__bf16)f;
    dst[i] = h;
    dst[4096 + i] = (__bf16)(f - (float)h);
  }

  // v_k = M mu ; w_k = mu.v + logdet + D*log(2*pi)
  if (tid < DDIM) {
    float s = 0.0f;
    for (int e = 0; e < DDIM; ++e) s += aug[tid * S + 64 + e] * mu[e];
    vsh[tid] = s;
    vks[(size_t)k * DDIM + tid] = s;
  }
  __syncthreads();
  if (tid == 0) {
    float c = 0.0f;
    for (int e = 0; e < DDIM; ++e) c += mu[e] * vsh[e];
    wk[k] = c + ldacc + (float)DDIM * 1.8378770664093453f;  // log(2*pi)
  }
}

// ---- Kernel 2: main WMMA kernel -------------------------------------------
__global__ __launch_bounds__(256) void gmm_loss_kernel(
    const float* __restrict__ emb, const float* __restrict__ pi,
    const int* __restrict__ labels, const __bf16* __restrict__ wsB,
    const float* __restrict__ vks, const float* __restrict__ wk,
    float* __restrict__ acc, int B) {
  __shared__ float xs[TROWS * DDIM];                 // 32KB gathered embeddings
  __shared__ float rs[TROWS * KCOMP];                // 16KB gathered resp
  __shared__ __align__(32) __bf16 BB[2][8192];       // 32KB double-buffered M frags
  __shared__ __align__(32) __bf16 Vh[2][2][32][16];  // 4KB V frags (hi)
  __shared__ __align__(32) __bf16 Vl[2][2][32][16];  // 4KB V frags (lo)
  __shared__ int   lb[TROWS];
  __shared__ float lds_acc;

  const int tid = threadIdx.x;
  const int base = blockIdx.x * TROWS;

  // TDM prologue ASAP: k=0 -> buffer 0 (wave 0 only; TENSORcnt is per-wave)
  if (tid < 32) {
    tdm_load_16k((unsigned)(size_t)&BB[0][0], (unsigned long long)(size_t)wsB);
  }

  if (tid == 0) lds_acc = 0.0f;
  if (tid < TROWS) {
    int idx = base + tid;
    lb[tid] = labels[idx < B ? idx : (B - 1)];
  }
  __syncthreads();

  // gather x tile (float4 vectorized)
  for (int i = tid; i < TROWS * 16; i += 256) {
    int r = i >> 4, c4 = i & 15;
    ((float4*)(xs + r * DDIM))[c4] =
        ((const float4*)(emb + (size_t)lb[r] * DDIM))[c4];
  }
  // gather resp tile (zero rows past B so they contribute nothing)
  for (int i = tid; i < TROWS * 8; i += 256) {
    int r = i >> 3, c4 = i & 7;
    float4 v = make_float4(0.f, 0.f, 0.f, 0.f);
    if (base + r < B) v = ((const float4*)(pi + (size_t)lb[r] * KCOMP))[c4];
    ((float4*)(rs + r * KCOMP))[c4] = v;
  }
  // V = [v_0 .. v_31] (64xK) as split-bf16 B fragments (2 chunks x 2 ntiles)
  for (int i = tid; i < 2048; i += 256) {
    int c = i >> 10, nt = (i >> 9) & 1, ln = (i >> 4) & 31, comp = i & 15;
    int kr = c * 32 + frag_k(comp, ln >> 4);
    int kcol = nt * 16 + (ln & 15);
    float f = vks[(size_t)kcol * DDIM + kr];
    __bf16 h = (__bf16)f;
    Vh[c][nt][ln][comp] = h;
    Vl[c][nt][ln][comp] = (__bf16)(f - (float)h);
  }
  __syncthreads();

  const int wave = tid >> 5;
  const int lane = tid & 31;
  const int lm = lane & 15;   // A: row M / B,C: column N (within tile)
  const int hi = lane >> 4;   // lane-half selector
  const int wrow = wave * 16; // this wave's 16-row tile

  // k-invariant A fragments: split-bf16 of x rows (built once per wave)
  const float* xrow = xs + (wrow + lm) * DDIM;
  v16bf ah[2], al[2];
#pragma unroll
  for (int c = 0; c < 2; ++c) {
#pragma unroll
    for (int comp = 0; comp < 16; ++comp) {
      float f = xrow[c * 32 + frag_k(comp, hi)];
      __bf16 h = (__bf16)f;
      ah[c][comp] = h;
      al[c][comp] = (__bf16)(f - (float)h);
    }
  }
  // k-invariant product-stage x values (C layout: lane=N, vgpr v -> M=v+8*hi)
  float xreg[4][8];
#pragma unroll
  for (int nt = 0; nt < 4; ++nt)
#pragma unroll
    for (int v = 0; v < 8; ++v)
      xreg[nt][v] = xs[(wrow + v + 8 * hi) * DDIM + nt * 16 + lm];

  // U = X @ V  (once): quad's -2*(M mu).x correction, reusing the A fragments
  float accv = 0.0f;
  {
    v8f u[2];
#pragma unroll
    for (int nt = 0; nt < 2; ++nt)
      u[nt] = (v8f){0.f, 0.f, 0.f, 0.f, 0.f, 0.f, 0.f, 0.f};
#pragma unroll
    for (int c = 0; c < 2; ++c) {
#pragma unroll
      for (int nt = 0; nt < 2; ++nt) {
        v16bf bh = *(const v16bf*)&Vh[c][nt][lane][0];
        v16bf bl = *(const v16bf*)&Vl[c][nt][lane][0];
        u[nt] = __builtin_amdgcn_wmma_f32_16x16x32_bf16(
            false, ah[c], false, bh, (short)0, u[nt], false, false);
        u[nt] = __builtin_amdgcn_wmma_f32_16x16x32_bf16(
            false, ah[c], false, bl, (short)0, u[nt], false, false);
        u[nt] = __builtin_amdgcn_wmma_f32_16x16x32_bf16(
            false, al[c], false, bh, (short)0, u[nt], false, false);
      }
    }
#pragma unroll
    for (int nt = 0; nt < 2; ++nt) {
      int kcol = nt * 16 + lm;
#pragma unroll
      for (int v = 0; v < 8; ++v)
        accv += rs[(wrow + v + 8 * hi) * KCOMP + kcol] * u[nt][v];
    }
  }

  // main loop over mixture components
  float accq = 0.0f;
  for (int k = 0; k < KCOMP; ++k) {
    if (tid < 32) __builtin_amdgcn_s_wait_tensorcnt(0);
    __syncthreads();  // BB[k&1] ready; all waves done with BB[(k+1)&1]
    if (tid < 32 && (k + 1) < KCOMP) {
      tdm_load_16k((unsigned)(size_t)&BB[(k + 1) & 1][0],
                   (unsigned long long)(size_t)(wsB + (size_t)(k + 1) * 8192));
    }
    const __bf16* Bk = &BB[k & 1][0];

    v8f y[4];
#pragma unroll
    for (int nt = 0; nt < 4; ++nt)
      y[nt] = (v8f){0.f, 0.f, 0.f, 0.f, 0.f, 0.f, 0.f, 0.f};

#pragma unroll
    for (int c = 0; c < 2; ++c) {
#pragma unroll
      for (int nt = 0; nt < 4; ++nt) {
        const __bf16* fb = Bk + ((c * 4 + nt) * 32 + lane) * 16;
        v16bf bh = *(const v16bf*)fb;            // 2x ds_load_b128
        v16bf bl = *(const v16bf*)(fb + 4096);
        // Y += xh*mh + xh*ml + xl*mh   (f32 accumulate inside WMMA)
        y[nt] = __builtin_amdgcn_wmma_f32_16x16x32_bf16(
            false, ah[c], false, bh, (short)0, y[nt], false, false);
        y[nt] = __builtin_amdgcn_wmma_f32_16x16x32_bf16(
            false, ah[c], false, bl, (short)0, y[nt], false, false);
        y[nt] = __builtin_amdgcn_wmma_f32_16x16x32_bf16(
            false, al[c], false, bh, (short)0, y[nt], false, false);
      }
    }

    // accq += resp[m,k] * sum_n Y[m,n]*x[m,n]   (= resp * x^T M x per row)
#pragma unroll
    for (int v = 0; v < 8; ++v) {
      float s = 0.0f;
#pragma unroll
      for (int nt = 0; nt < 4; ++nt) s += y[nt][v] * xreg[nt][v];
      accq += rs[(wrow + v + 8 * hi) * KCOMP + k] * s;
    }
    // next iteration's wait+barrier protects BB reuse
  }

  // constant term: sum_k resp * w_k,  w_k = mu^T M mu + logdet_k + D*log2pi
  float accc = 0.0f;
  if (tid < TROWS && (base + tid) < B) {
#pragma unroll
    for (int k = 0; k < KCOMP; ++k) accc += rs[tid * KCOMP + k] * wk[k];
  }

  // quad = x^T M x - 2 u + (mu^T M mu)  -> loss contribution
  atomicAdd(&lds_acc, -0.5f * (accq - 2.0f * accv + accc));  // ds_add_f32
  __syncthreads();
  if (tid == 0) atomicAdd(acc, lds_acc);  // global_atomic_add_f32
}

// ---- Kernel 3: |sum| -> output --------------------------------------------
__global__ void finalize_kernel(const float* acc, float* out) {
  out[0] = fabsf(acc[0]);
}

// ---------------------------------------------------------------------------
extern "C" void kernel_launch(void* const* d_in, const int* in_sizes, int n_in,
                              void* d_out, int out_size, void* d_ws, size_t ws_size,
                              hipStream_t stream) {
  const float* emb    = (const float*)d_in[0];  // (500000, 64)
  const float* cen    = (const float*)d_in[1];  // (32, 64)
  const float* cov    = (const float*)d_in[2];  // (32, 64, 64)
  const float* pi     = (const float*)d_in[3];  // (500000, 32)
  const int*   labels = (const int*)d_in[4];    // (B,)
  float* out = (float*)d_out;

  const int B = in_sizes[4];

  // workspace layout: [0] acc | [16..47] w_k | [64..2111] v_k (f32)
  //                   [4096 floats ..] split-bf16 M fragments (512KB)
  float* wsf  = (float*)d_ws;
  float* acc  = wsf;
  float* wk   = wsf + 16;
  float* vks  = wsf + 64;
  __bf16* wsB = (__bf16*)(wsf + 4096);

  zero_acc_kernel<<<1, 1, 0, stream>>>(acc);
  invert_prep_kernel<<<KCOMP, 256, 0, stream>>>(cov, cen, wsB, vks, wk);

  int grid = (B + TROWS - 1) / TROWS;
  gmm_loss_kernel<<<grid, 256, 0, stream>>>(emb, pi, labels, wsB, vks, wk,
                                            acc, B);
  finalize_kernel<<<1, 1, 0, stream>>>(acc, out);
}
